// SymAgg_2929167696454
// MI455X (gfx1250) — matrix operand order repaired and checked by reading, not compiled, not run on hardware
//
#include <hip/hip_runtime.h>
#include <hip/hip_bf16.h>

#define DEVINL __device__ __forceinline__

constexpr int Bn = 2048;   // batch
constexpr int Sn = 8;      // num_sym
constexpr int Nd = 4096;   // n_disease
constexpr int Hn = 256;    // hidden
constexpr int BS = Bn * Sn;

typedef __attribute__((ext_vector_type(16))) __bf16       v16bf;
typedef __attribute__((ext_vector_type(8)))  float        v8f;
typedef __attribute__((ext_vector_type(8)))  unsigned int v8u;

DEVINL unsigned short f2bf(float f) {
  unsigned int u = __builtin_bit_cast(unsigned int, f);
  u += 0x7FFFu + ((u >> 16) & 1u);           // round-to-nearest-even
  return (unsigned short)(u >> 16);
}

DEVINL v16bf pack_frag(uint4 a, uint4 b) {
  v8u v;
  v[0] = a.x; v[1] = a.y; v[2] = a.z; v[3] = a.w;
  v[4] = b.x; v[5] = b.y; v[6] = b.z; v[7] = b.w;
  return __builtin_bit_cast(v16bf, v);
}

// A fragment (16x32 bf16): lane&15 -> row, lane>>4 selects K-half.
// VGPR0-3 = K [hs*16, hs*16+7], VGPR4-7 = K [hs*16+8, hs*16+15]
DEVINL v16bf load_a_frag(const unsigned short* __restrict__ rowptr, int k0, int hs) {
  uint4 lo = *(const uint4*)(rowptr + k0 + hs * 16);
  uint4 hi = *(const uint4*)(rowptr + k0 + hs * 16 + 8);
  return pack_frag(lo, hi);
}

// B fragment (32x16 bf16) from a K-contiguous-per-column source:
// lanes 0-15 hold K = k0..k0+15 of column lane&15, lanes 16-31 hold K = k0+16..k0+31
DEVINL v16bf load_b_frag(const unsigned short* __restrict__ colptr, int k0, int hs) {
  uint4 lo = *(const uint4*)(colptr + k0 + hs * 16);
  uint4 hi = *(const uint4*)(colptr + k0 + hs * 16 + 8);
  return pack_frag(lo, hi);
}

DEVINL v8f wmma_bf16(v16bf a, v16bf b, v8f c) {
  return __builtin_amdgcn_wmma_f32_16x16x32_bf16(false, a, false, b, (short)0, c,
                                                 false, false);
}

// ---- CDNA5 async Global->LDS copy (one wave instruction, 16B per lane) ----
DEVINL void async_copy_b128(unsigned short* lds_dst, const unsigned short* gsrc) {
#if defined(__gfx1250__)
  auto lp = (__attribute__((address_space(3))) unsigned short*)lds_dst;
  unsigned long long ga = (unsigned long long)(uintptr_t)gsrc;
  asm volatile("global_load_async_to_lds_b128 %0, %1, off"
               :: "v"(lp), "v"(ga) : "memory");
#else
  *(uint4*)lds_dst = *(const uint4*)gsrc;
#endif
}

DEVINL void wait_async0() {
#if defined(__gfx1250__)
#if __has_builtin(__builtin_amdgcn_s_wait_asynccnt)
  __builtin_amdgcn_s_wait_asynccnt(0);
#else
  asm volatile("s_wait_asynccnt 0x0" ::: "memory");
#endif
#endif
}

// Shared 16x64 wave-tile inner loop (direct-global B), used by the 256-K GEMMs.
DEVINL void gemm_core_k(const unsigned short* __restrict__ arow,
                        const unsigned short* const bcol[4],
                        int kdim, int hs, v8f acc[4]) {
  for (int k0 = 0; k0 < kdim; k0 += 32) {
    v16bf a = load_a_frag(arow, k0, hs);
#pragma unroll
    for (int t = 0; t < 4; ++t) {
      v16bf b = load_b_frag(bcol[t], k0, hs);
      acc[t] = wmma_bf16(a, b, acc[t]);
    }
  }
}

// ---------------- stage kernels ----------------

// scores[s,n] = dot(E[s,n,:], att)   (one wave per row)
__global__ void __launch_bounds__(256) k_scores(const float* __restrict__ E,
                                                const float* __restrict__ att,
                                                float* __restrict__ scores) {
  int gw   = (int)((blockIdx.x * blockDim.x + threadIdx.x) >> 5);
  int lane = threadIdx.x & 31;
  if (gw >= Sn * Nd) return;
  const float* row = E + (size_t)gw * Hn;
  float acc = 0.f;
  for (int h = lane; h < Hn; h += 32) acc += row[h] * att[h];
#pragma unroll
  for (int o = 16; o; o >>= 1) acc += __shfl_xor(acc, o);
  if (lane == 0) scores[gw] = acc;
}

// per-s global max (safe upper bound for the masked softmax shift)
__global__ void __launch_bounds__(256) k_max(const float* __restrict__ scores,
                                             float* __restrict__ maxs) {
  __shared__ float red[256];
  int s = blockIdx.x;
  float m = -3.4e38f;
  for (int n = threadIdx.x; n < Nd; n += 256) m = fmaxf(m, scores[(size_t)s * Nd + n]);
  red[threadIdx.x] = m;
  __syncthreads();
  for (int w = 128; w; w >>= 1) {
    if ((int)threadIdx.x < w) red[threadIdx.x] = fmaxf(red[threadIdx.x], red[threadIdx.x + w]);
    __syncthreads();
  }
  if (threadIdx.x == 0) maxs[s] = red[0];
}

__global__ void __launch_bounds__(256) k_exp(const float* __restrict__ scores,
                                             const float* __restrict__ maxs,
                                             float* __restrict__ p) {
  int i = blockIdx.x * 256 + threadIdx.x;
  if (i < Sn * Nd) p[i] = expf(scores[i] - maxs[i / Nd]);
}

// XbfT[s,h,n] = bf16(p[s,n] * E[s,n,h])  -- LDS 32x32 tile transpose
__global__ void __launch_bounds__(256) k_prep(const float* __restrict__ E,
                                              const float* __restrict__ p,
                                              unsigned short* __restrict__ XbfT) {
  __shared__ float tile[32][33];
  int s = blockIdx.z, n0 = blockIdx.x * 32, h0 = blockIdx.y * 32;
  int tx = threadIdx.x, ty = threadIdx.y;
  const float* Es = E + (size_t)s * Nd * Hn;
#pragma unroll
  for (int k = 0; k < 4; ++k)
    tile[ty + k * 8][tx] = Es[(size_t)(n0 + ty + k * 8) * Hn + h0 + tx];
  __syncthreads();
  float pv = p[(size_t)s * Nd + n0 + tx];
#pragma unroll
  for (int k = 0; k < 4; ++k) {
    int h = h0 + ty + k * 8;
    XbfT[((size_t)s * Hn + h) * Nd + n0 + tx] = f2bf(tile[tx][ty + k * 8] * pv);
  }
}

__global__ void __launch_bounds__(256) k_cvt_i2bf(const int* __restrict__ src,
                                                  unsigned short* __restrict__ dst, int n) {
  int i = blockIdx.x * 256 + threadIdx.x;
  if (i < n) dst[i] = f2bf((float)src[i]);
}

__global__ void __launch_bounds__(256) k_cvt_f2bf(const float* __restrict__ src,
                                                  unsigned short* __restrict__ dst, int n) {
  int i = blockIdx.x * 256 + threadIdx.x;
  if (i < n) dst[i] = f2bf(src[i]);
}

// denom[b,s] = sum_n mask[b,n] * p[s,n]   (one wave per b, all 8 s at once)
__global__ void __launch_bounds__(256) k_denom(const int* __restrict__ dm,
                                               const float* __restrict__ p,
                                               float* __restrict__ denom) {
  int b    = (int)((blockIdx.x * blockDim.x + threadIdx.x) >> 5);
  int lane = threadIdx.x & 31;
  const int* drow = dm + (size_t)b * Nd;
  float acc[Sn] = {};
  for (int n = lane; n < Nd; n += 32) {
    float m = (float)drow[n];
#pragma unroll
    for (int s = 0; s < Sn; ++s) acc[s] += m * p[(size_t)s * Nd + n];
  }
#pragma unroll
  for (int s = 0; s < Sn; ++s) {
    float v = acc[s];
#pragma unroll
    for (int o = 16; o; o >>= 1) v += __shfl_xor(v, o);
    if (lane == 0) denom[(size_t)b * Sn + s] = v;
  }
}

// b_sym (bf16) = (mask @ (p*E)) / denom : per-s GEMM 2048x4096x256.
// B tile (32K x 64N, 4KB) staged via async Global->LDS, double-buffered:
// each of the 256 threads issues exactly one global_load_async_to_lds_b128
// per K-step; waves read fragments back with ds_load_b128.
__global__ void __launch_bounds__(256) k_gemm_bsym(const unsigned short* __restrict__ Abf,
                                                   const unsigned short* __restrict__ XbfT,
                                                   const float* __restrict__ denom,
                                                   unsigned short* __restrict__ bsym) {
  __shared__ unsigned short ldsB[2][64 * 32];  // [buf][col*32 + k]
  int s = blockIdx.z;
  int tid = threadIdx.x;
  int wave = tid >> 5, lane = tid & 31;
  int hs = lane >> 4, l16 = lane & 15;
  int m0 = blockIdx.x * 128 + wave * 16;
  int n0 = blockIdx.y * 64;
  const unsigned short* arow = Abf + (size_t)(m0 + l16) * Nd;
  const unsigned short* bbase = XbfT + (size_t)s * Hn * Nd;

  // async-copy mapping: thread -> (column, 8-elem chunk) of the 32x64 tile
  int acol = tid >> 2;             // 0..63
  int achk = (tid & 3) * 8;        // 0,8,16,24 (elements within K)
  const unsigned short* gsrc = bbase + (size_t)(n0 + acol) * Nd + achk;
  unsigned short* ldst0 = &ldsB[0][acol * 32 + achk];
  unsigned short* ldst1 = &ldsB[1][acol * 32 + achk];

  async_copy_b128(ldst0, gsrc);    // prologue: fill buffer 0 (k0 = 0)
  wait_async0();
  __syncthreads();

  v8f acc[4] = {};
  for (int k0 = 0; k0 < Nd; k0 += 32) {
    int cur = (k0 >> 5) & 1;
    if (k0 + 32 < Nd)              // prefetch next tile into the other buffer
      async_copy_b128(cur ? ldst0 : ldst1, gsrc + k0 + 32);
    v16bf a = load_a_frag(arow, k0, hs);
#pragma unroll
    for (int t = 0; t < 4; ++t) {
      const unsigned short* cp = &ldsB[cur][(t * 16 + l16) * 32 + hs * 16];
      uint4 lo = *(const uint4*)cp;
      uint4 hi = *(const uint4*)(cp + 8);
      acc[t] = wmma_bf16(a, pack_frag(lo, hi), acc[t]);
    }
    wait_async0();                 // my async writes for the next buffer done
    __syncthreads();               // everyone's writes done / reads of cur done
  }

#pragma unroll
  for (int t = 0; t < 4; ++t) {
    int col = n0 + t * 16 + l16;
#pragma unroll
    for (int i = 0; i < 8; ++i) {
      int row = m0 + i + hs * 8;
      float d = denom[(size_t)row * Sn + s];
      bsym[((size_t)row * Sn + s) * Hn + col] = f2bf(acc[t][i] / d);
    }
  }
}

// v (bf16) = bsym @ Wv^T + bv
__global__ void __launch_bounds__(256) k_gemm_v(const unsigned short* __restrict__ A,
                                                const unsigned short* __restrict__ W,
                                                const float* __restrict__ bias,
                                                unsigned short* __restrict__ out) {
  int wave = threadIdx.x >> 5, lane = threadIdx.x & 31;
  int hs = lane >> 4, l16 = lane & 15;
  int m0 = blockIdx.x * 128 + wave * 16;
  int n0 = blockIdx.y * 64;
  const unsigned short* arow = A + (size_t)(m0 + l16) * Hn;
  const unsigned short* bcol[4];
#pragma unroll
  for (int t = 0; t < 4; ++t) bcol[t] = W + (size_t)(n0 + t * 16 + l16) * Hn;
  v8f acc[4] = {};
  gemm_core_k(arow, bcol, Hn, hs, acc);
#pragma unroll
  for (int t = 0; t < 4; ++t) {
    int col = n0 + t * 16 + l16;
    float bc = bias[col];
#pragma unroll
    for (int i = 0; i < 8; ++i) {
      int row = m0 + i + hs * 8;
      out[(size_t)row * Hn + col] = f2bf(acc[t][i] + bc);
    }
  }
}

// sdpa = v @ Wo^T + bo  (store f32 + bf16)
__global__ void __launch_bounds__(256) k_gemm_sdpa(const unsigned short* __restrict__ A,
                                                   const unsigned short* __restrict__ W,
                                                   const float* __restrict__ bias,
                                                   float* __restrict__ outf,
                                                   unsigned short* __restrict__ outb) {
  int wave = threadIdx.x >> 5, lane = threadIdx.x & 31;
  int hs = lane >> 4, l16 = lane & 15;
  int m0 = blockIdx.x * 128 + wave * 16;
  int n0 = blockIdx.y * 64;
  const unsigned short* arow = A + (size_t)(m0 + l16) * Hn;
  const unsigned short* bcol[4];
#pragma unroll
  for (int t = 0; t < 4; ++t) bcol[t] = W + (size_t)(n0 + t * 16 + l16) * Hn;
  v8f acc[4] = {};
  gemm_core_k(arow, bcol, Hn, hs, acc);
#pragma unroll
  for (int t = 0; t < 4; ++t) {
    int col = n0 + t * 16 + l16;
    float bc = bias[col];
#pragma unroll
    for (int i = 0; i < 8; ++i) {
      size_t idx = (size_t)(m0 + i + hs * 8) * Hn + col;
      float v = acc[t][i] + bc;
      outf[idx] = v;
      outb[idx] = f2bf(v);
    }
  }
}

// b_sym_syn = sdpa + sdpa @ lin_w[s]^T + lin_b[s]  (per-s, s-strided A rows)
__global__ void __launch_bounds__(256) k_gemm_lin(const unsigned short* __restrict__ sdpab,
                                                  const float* __restrict__ sdpaf,
                                                  const unsigned short* __restrict__ linw,
                                                  const float* __restrict__ linb,
                                                  float* __restrict__ outf,
                                                  unsigned short* __restrict__ outb) {
  int s = blockIdx.z;
  int wave = threadIdx.x >> 5, lane = threadIdx.x & 31;
  int hs = lane >> 4, l16 = lane & 15;
  int m0 = blockIdx.x * 128 + wave * 16;   // batch row
  int n0 = blockIdx.y * 64;
  const unsigned short* arow = sdpab + ((size_t)(m0 + l16) * Sn + s) * Hn;
  const unsigned short* bbase = linw + (size_t)s * Hn * Hn;
  const unsigned short* bcol[4];
#pragma unroll
  for (int t = 0; t < 4; ++t) bcol[t] = bbase + (size_t)(n0 + t * 16 + l16) * Hn;
  v8f acc[4] = {};
  gemm_core_k(arow, bcol, Hn, hs, acc);
#pragma unroll
  for (int t = 0; t < 4; ++t) {
    int col = n0 + t * 16 + l16;
    float bc = linb[(size_t)s * Hn + col];
#pragma unroll
    for (int i = 0; i < 8; ++i) {
      int b = m0 + i + hs * 8;
      size_t idx = ((size_t)b * Sn + s) * Hn + col;
      float v = sdpaf[idx] + acc[t][i] + bc;
      outf[idx] = v;
      outb[idx] = f2bf(v);
    }
  }
}

// t = tanh(bss @ sym_W1^T + sym_b1)
__global__ void __launch_bounds__(256) k_gemm_tanh(const unsigned short* __restrict__ A,
                                                   const unsigned short* __restrict__ W,
                                                   const float* __restrict__ bias,
                                                   float* __restrict__ outf) {
  int wave = threadIdx.x >> 5, lane = threadIdx.x & 31;
  int hs = lane >> 4, l16 = lane & 15;
  int m0 = blockIdx.x * 128 + wave * 16;
  int n0 = blockIdx.y * 64;
  const unsigned short* arow = A + (size_t)(m0 + l16) * Hn;
  const unsigned short* bcol[4];
#pragma unroll
  for (int t = 0; t < 4; ++t) bcol[t] = W + (size_t)(n0 + t * 16 + l16) * Hn;
  v8f acc[4] = {};
  gemm_core_k(arow, bcol, Hn, hs, acc);
#pragma unroll
  for (int t = 0; t < 4; ++t) {
    int col = n0 + t * 16 + l16;
    float bc = bias[col];
#pragma unroll
    for (int i = 0; i < 8; ++i)
      outf[(size_t)(m0 + i + hs * 8) * Hn + col] = tanhf(acc[t][i] + bc);
  }
}

// e2[r] = dot(t[r,:], sym_w2)
__global__ void __launch_bounds__(256) k_e2(const float* __restrict__ tf,
                                            const float* __restrict__ w2,
                                            float* __restrict__ e2) {
  int gw   = (int)((blockIdx.x * blockDim.x + threadIdx.x) >> 5);
  int lane = threadIdx.x & 31;
  const float* row = tf + (size_t)gw * Hn;
  float acc = 0.f;
  for (int h = lane; h < Hn; h += 32) acc += row[h] * w2[h];
#pragma unroll
  for (int o = 16; o; o >>= 1) acc += __shfl_xor(acc, o);
  if (lane == 0) e2[gw] = acc;
}

// softmax over S (=8) and pool:  out[b,h] = sum_s beta2[b,s] * bss[b,s,h]
__global__ void __launch_bounds__(256) k_pool(const float* __restrict__ e2,
                                              const float* __restrict__ bss,
                                              float* __restrict__ out) {
  int b = blockIdx.x;
  int h = threadIdx.x;
  float e[Sn], m = -3.4e38f;
#pragma unroll
  for (int s = 0; s < Sn; ++s) { e[s] = e2[(size_t)b * Sn + s]; m = fmaxf(m, e[s]); }
  float den = 0.f, acc = 0.f;
#pragma unroll
  for (int s = 0; s < Sn; ++s) {
    float w = expf(e[s] - m);
    den += w;
    acc += w * bss[((size_t)b * Sn + s) * Hn + h];
  }
  out[(size_t)b * Hn + h] = acc / den;
}

// ---------------- host launcher ----------------

extern "C" void kernel_launch(void* const* d_in, const int* in_sizes, int n_in,
                              void* d_out, int out_size, void* d_ws, size_t ws_size,
                              hipStream_t stream) {
  const float* E     = (const float*)d_in[0];   // [S,N,H]
  const int*   dmask = (const int*)  d_in[2];   // [B,N]
  const float* att   = (const float*)d_in[4];   // [1,H]
  const float* symW1 = (const float*)d_in[8];   // [H,H]
  const float* symb1 = (const float*)d_in[9];
  const float* symw2 = (const float*)d_in[10];
  const float* Wv    = (const float*)d_in[15];
  const float* bv    = (const float*)d_in[16];
  const float* Wo    = (const float*)d_in[17];
  const float* bo    = (const float*)d_in[18];
  const float* linw  = (const float*)d_in[19];  // [S,H,H]
  const float* linb  = (const float*)d_in[20];  // [S,H]
  float* out = (float*)d_out;                   // [B,1,H]

  size_t off = 0;
  auto alloc = [&](size_t bytes) -> void* {
    void* p = (char*)d_ws + off;
    off += (bytes + 255) & ~(size_t)255;
    return p;
  };
  float* scores = (float*)alloc((size_t)Sn * Nd * 4);
  float* maxs   = (float*)alloc(256);
  float* pbuf   = (float*)alloc((size_t)Sn * Nd * 4);
  float* denom  = (float*)alloc((size_t)Bn * Sn * 4);
  float* e2     = (float*)alloc((size_t)BS * 4);
  unsigned short* wv_bf  = (unsigned short*)alloc((size_t)Hn * Hn * 2);
  unsigned short* wo_bf  = (unsigned short*)alloc((size_t)Hn * Hn * 2);
  unsigned short* w1_bf  = (unsigned short*)alloc((size_t)Hn * Hn * 2);
  unsigned short* lin_bf = (unsigned short*)alloc((size_t)Sn * Hn * Hn * 2);
  unsigned short* Abf    = (unsigned short*)alloc((size_t)Bn * Nd * 2);      // 16 MB
  unsigned short* XbfT   = (unsigned short*)alloc((size_t)Sn * Hn * Nd * 2); // 16 MB
  unsigned short* bsym   = (unsigned short*)alloc((size_t)BS * Hn * 2);
  unsigned short* vbf    = (unsigned short*)alloc((size_t)BS * Hn * 2);
  unsigned short* sdpab  = (unsigned short*)alloc((size_t)BS * Hn * 2);
  float*          bssf   = (float*)alloc((size_t)BS * Hn * 4);
  unsigned short* bssb   = (unsigned short*)alloc((size_t)BS * Hn * 2);
  // dead-buffer aliases (equal sizes, producers run strictly after consumers):
  float* sdpaf = (float*)XbfT;  // XbfT consumed by k_gemm_bsym before k_gemm_sdpa writes
  float* tf    = (float*)Abf;   // Abf consumed by k_gemm_bsym before k_gemm_tanh writes

  dim3 blk(256);
  // 1) scores, max, exp
  k_scores<<<dim3((Sn * Nd) / 8), blk, 0, stream>>>(E, att, scores);
  k_max<<<dim3(Sn), blk, 0, stream>>>(scores, maxs);
  k_exp<<<dim3((Sn * Nd + 255) / 256), blk, 0, stream>>>(scores, maxs, pbuf);
  // 2) operand prep
  k_prep<<<dim3(Nd / 32, Hn / 32, Sn), dim3(32, 8), 0, stream>>>(E, pbuf, XbfT);
  k_cvt_i2bf<<<dim3(((size_t)Bn * Nd + 255) / 256), blk, 0, stream>>>(dmask, Abf, Bn * Nd);
  k_denom<<<dim3(Bn / 8), blk, 0, stream>>>(dmask, pbuf, denom);
  k_cvt_f2bf<<<dim3((Hn * Hn + 255) / 256), blk, 0, stream>>>(Wv, wv_bf, Hn * Hn);
  k_cvt_f2bf<<<dim3((Hn * Hn + 255) / 256), blk, 0, stream>>>(Wo, wo_bf, Hn * Hn);
  k_cvt_f2bf<<<dim3((Hn * Hn + 255) / 256), blk, 0, stream>>>(symW1, w1_bf, Hn * Hn);
  k_cvt_f2bf<<<dim3((Sn * Hn * Hn + 255) / 256), blk, 0, stream>>>(linw, lin_bf, Sn * Hn * Hn);
  // 3) dominant GEMM: b_sym = (mask @ p*E) / denom   [8 x (2048x4096x256)]
  //    async Global->LDS double-buffered B tiles + WMMA
  k_gemm_bsym<<<dim3(Bn / 128, Hn / 64, Sn), blk, 0, stream>>>(Abf, XbfT, denom, bsym);
  // 4) projection chain (att == 1 makes q/k dead)
  k_gemm_v<<<dim3(BS / 128, Hn / 64), blk, 0, stream>>>(bsym, wv_bf, bv, vbf);
  k_gemm_sdpa<<<dim3(BS / 128, Hn / 64), blk, 0, stream>>>(vbf, wo_bf, bo, sdpaf, sdpab);
  k_gemm_lin<<<dim3(Bn / 128, Hn / 64, Sn), blk, 0, stream>>>(sdpab, sdpaf, lin_bf, linb,
                                                              bssf, bssb);
  // 5) pooling attention over S
  k_gemm_tanh<<<dim3(BS / 128, Hn / 64), blk, 0, stream>>>(bssb, w1_bf, symb1, tf);
  k_e2<<<dim3(BS / 8), blk, 0, stream>>>(tf, symw2, e2);
  k_pool<<<dim3(Bn), blk, 0, stream>>>(e2, bssf, out);
  (void)in_sizes; (void)n_in; (void)out_size; (void)ws_size;
}
